// Attention_36180804501950
// MI455X (gfx1250) — matrix operand order repaired and checked by reading, not compiled, not run on hardware
//
#include <hip/hip_runtime.h>
#include <hip/hip_bf16.h>

#define SEQ   2048
#define DIMN  1024
#define HEADS 16
#define HD    64
#define QKH   6
#define KKW   11
#define EPSV  1e-5f

typedef __attribute__((ext_vector_type(16))) __bf16   v16bf;
typedef __attribute__((ext_vector_type(8)))  float    v8f;
typedef __attribute__((ext_vector_type(4)))  unsigned uu4;
typedef __attribute__((ext_vector_type(8)))  unsigned uu8;

union Frag { unsigned u[8]; uu4 q[2]; v16bf v; };

__device__ inline __bf16 tobf(float f) {
    unsigned u = __builtin_bit_cast(unsigned, f);
    u += 0x7FFFu + ((u >> 16) & 1u);
    unsigned short h = (unsigned short)(u >> 16);
    return __builtin_bit_cast(__bf16, h);
}

// A fragment: 16x32 bf16, row-major, ISA layout (v0..3: K=hi*8+2j ; v4..7: K=16+hi*8+..)
__device__ inline void loadA(Frag& f, const __bf16* __restrict__ A, int lda,
                             int r0, int kb, int lane) {
    int m = lane & 15, hi = lane >> 4;
    const __bf16* p = A + (size_t)(r0 + m) * lda + kb + hi * 8;
    f.q[0] = *(const uu4*)p;
    f.q[1] = *(const uu4*)(p + 16);
}

// B fragment: 32x16 bf16 from Bt stored [N][K], ISA layout (v0..3: K=hi*8+.. ; v4..7: K=16+hi*8+..)
__device__ inline void loadB(Frag& f, const __bf16* __restrict__ Bt, int ldb,
                             int n0, int kb, int lane) {
    int n = lane & 15, hi = lane >> 4;
    const __bf16* p = Bt + (size_t)(n0 + n) * ldb + kb + hi * 8;
    f.q[0] = *(const uu4*)p;
    f.q[1] = *(const uu4*)(p + 16);
}

__device__ inline v8f wmma_bf16(const Frag& a, const Frag& b, v8f c) {
    return __builtin_amdgcn_wmma_f32_16x16x32_bf16(
        false, a.v, false, b.v, (short)0, c, false, false);
}

// ---------------- elementwise converts ----------------
__global__ void cvt_bf16_kernel(const float* __restrict__ src,
                                __bf16* __restrict__ dst, int n) {
    int i = blockIdx.x * blockDim.x + threadIdx.x;
    if (i < n) dst[i] = tobf(src[i]);
}

__global__ void transpose_bf16_kernel(const float* __restrict__ src,
                                      __bf16* __restrict__ dst, int K, int N) {
    int i = blockIdx.x * blockDim.x + threadIdx.x;
    if (i >= K * N) return;
    int n = i / K, k = i % K;
    dst[i] = tobf(src[(size_t)k * N + n]);
}

// -------- generic WMMA GEMM: C[M,N] = A[M,K] * Bt[N,K]^T, 32x64 tile per wave --------
__global__ __launch_bounds__(256) void gemm_bf16_nt(
    const __bf16* __restrict__ A, int lda,
    const __bf16* __restrict__ Bt, int ldb,
    float* __restrict__ C, int ldc,
    int M, int N, int K)
{
    int wid  = (int)((blockIdx.x * blockDim.x + threadIdx.x) >> 5);
    int lane = threadIdx.x & 31;
    int ngrp = N >> 6;
    int mt = wid / ngrp, nb = wid % ngrp;
    int r0 = mt << 5;                       // 2 M-tiles per wave
    if (r0 >= M) return;                    // wave-uniform
    int n0 = nb << 6;

    v8f acc[2][4] = {};
    for (int kb = 0; kb < K; kb += 32) {
        Frag a0, a1;
        loadA(a0, A, lda, r0,      kb, lane);
        loadA(a1, A, lda, r0 + 16, kb, lane);
        if (kb + 32 < K) {
            __builtin_prefetch(A + (size_t)(r0 + (lane & 15)) * lda + kb + 32, 0, 1);
            __builtin_prefetch(Bt + (size_t)(n0 + (lane & 15)) * ldb + kb + 32, 0, 1);
        }
#pragma unroll
        for (int t = 0; t < 4; t++) {
            Frag b; loadB(b, Bt, ldb, n0 + t * 16, kb, lane);
            acc[0][t] = wmma_bf16(a0, b, acc[0][t]);
            acc[1][t] = wmma_bf16(a1, b, acc[1][t]);
        }
    }
    int nn = lane & 15, hi = lane >> 4;
#pragma unroll
    for (int half = 0; half < 2; half++)
#pragma unroll
        for (int t = 0; t < 4; t++)
#pragma unroll
            for (int j = 0; j < 8; j++)
                C[(size_t)(r0 + half * 16 + j + hi * 8) * ldc + n0 + t * 16 + nn] =
                    acc[half][t][j];
}

// ---------------- RoPE + repack into attention layouts ----------------
__global__ void rope_pack_kernel(const float* __restrict__ qf,
                                 const float* __restrict__ kf,
                                 const float* __restrict__ vf,
                                 const float* __restrict__ freq,
                                 __bf16* __restrict__ qb,
                                 __bf16* __restrict__ kb,
                                 __bf16* __restrict__ vt)
{
    int idx = blockIdx.x * blockDim.x + threadIdx.x;   // SEQ*HEADS*32 threads
    int p = idx & 31;
    int h = (idx >> 5) & (HEADS - 1);
    int s = idx >> 9;
    if (s >= SEQ) return;
    float c  = freq[((size_t)s * 32 + p) * 2 + 0];
    float sn = freq[((size_t)s * 32 + p) * 2 + 1];
    size_t src = (size_t)s * DIMN + h * HD + 2 * p;
    float q0 = qf[src], q1 = qf[src + 1];
    float k0 = kf[src], k1 = kf[src + 1];
    float qr = q0 * c - q1 * sn, qi = q0 * sn + q1 * c;
    float kr = k0 * c - k1 * sn, ki = k0 * sn + k1 * c;
    size_t dst = (size_t)h * SEQ * HD + (size_t)s * HD + 2 * p;
    qb[dst] = tobf(qr); qb[dst + 1] = tobf(qi);
    kb[dst] = tobf(kr); kb[dst + 1] = tobf(ki);
    float v0 = vf[src], v1 = vf[src + 1];
    size_t vd = (size_t)h * HD * SEQ + (size_t)(2 * p) * SEQ + s;
    vt[vd] = tobf(v0); vt[vd + SEQ] = tobf(v1);
}

// ---------------- masked scores: sc[q][k] = (k<=q) ? scale * q.k : 0 ----------------
__global__ __launch_bounds__(256) void scores_kernel(
    const __bf16* __restrict__ qh, const __bf16* __restrict__ kh,
    float* __restrict__ sc, float scale)
{
    int wid  = (int)((blockIdx.x * blockDim.x + threadIdx.x) >> 5);
    int lane = threadIdx.x & 31;
    const int KG = SEQ / 64;
    int qt = wid / KG, kg = wid % KG;
    int q0 = qt * 16, k0 = kg * 64;
    int nn = lane & 15, hi = lane >> 4;

    if (k0 > q0 + 15) {
#pragma unroll
        for (int t = 0; t < 4; t++)
#pragma unroll
            for (int j = 0; j < 8; j++)
                sc[(size_t)(q0 + j + hi * 8) * SEQ + k0 + t * 16 + nn] = 0.f;
        return;
    }
    v8f acc[4] = {};
#pragma unroll
    for (int kb = 0; kb < HD; kb += 32) {
        Frag a; loadA(a, qh, HD, q0, kb, lane);
#pragma unroll
        for (int t = 0; t < 4; t++) {
            Frag b; loadB(b, kh, HD, k0 + t * 16, kb, lane);
            acc[t] = wmma_bf16(a, b, acc[t]);
        }
    }
#pragma unroll
    for (int t = 0; t < 4; t++)
#pragma unroll
        for (int j = 0; j < 8; j++) {
            int q = q0 + j + hi * 8;
            int k = k0 + t * 16 + nn;
            sc[(size_t)q * SEQ + k] = (k <= q) ? acc[t][j] * scale : 0.f;
        }
}

// ------- conv (6x11) + causal softmax; score rows staged into LDS by the TDM -------
#define SWPAD (SEQ + 32)     // 8 left halo, interior [8, 8+SEQ), right halo to 2080

// Issue one 1-row tensor_load_to_lds (2D descriptor, groups 2/3 null).
__device__ inline void tdm_load_row(unsigned lds_byte_off, const float* gsrc) {
    unsigned long long ga = (unsigned long long)(size_t)gsrc;
    uu4 g0;
    g0[0] = 1u;                                    // count=1, user mode
    g0[1] = lds_byte_off;                          // lds_addr
    g0[2] = (unsigned)(ga & 0xffffffffull);        // global_addr[31:0]
    g0[3] = (unsigned)((ga >> 32) & 0x01ffffffu) | (2u << 30);  // addr[56:32], type=2
    uu8 g1;
    g1[0] = 2u << 16;                              // data_size = 4 bytes
    g1[1] = (SEQ & 0xffffu) << 16;                 // tensor_dim0[15:0] = 2048
    g1[2] = 1u << 16;                              // tensor_dim0 hi=0, tensor_dim1 = 1
    g1[3] = (SEQ & 0xffffu) << 16;                 // tile_dim0 = 2048
    g1[4] = 1u;                                    // tile_dim1 = 1, tile_dim2 = 0
    g1[5] = (unsigned)SEQ;                         // tensor_dim0_stride = 2048
    g1[6] = 0u;
    g1[7] = 0u;
    asm volatile("tensor_load_to_lds %0, %1, null, null"
                 :: "s"(g0), "s"(g1) : "memory");
}

__global__ __launch_bounds__(256) void conv_softmax_kernel(
    const float* __restrict__ sc, const float* __restrict__ kern,
    __bf16* __restrict__ probs)
{
    __shared__ float rows[QKH][SWPAD];
    __shared__ float conv[SEQ];
    __shared__ float red[256];
    __shared__ float kk[QKH * KKW];
    int q = blockIdx.x, tid = threadIdx.x;
    if (tid < QKH * KKW) kk[tid] = kern[tid];

    // zero halos / invalid rows (disjoint from the TDM interior writes)
#pragma unroll
    for (int i = 0; i < QKH; i++) {
        int qr = q - (QKH - 1) + i;
        if (qr < 0) {
            for (int c = tid; c < SWPAD; c += 256) rows[i][c] = 0.f;
        } else {
            if (tid < 8)                 rows[i][tid] = 0.f;          // left halo
            else if (tid < 8 + 24)       rows[i][SEQ + tid] = 0.f;    // right halo [2056,2080)
        }
    }
    // wave 0 issues the TDM loads for the valid rows; TDM ignores EXEC, 1 issue/wave
    if (threadIdx.x < 32) {
#pragma unroll
        for (int i = 0; i < QKH; i++) {
            int qr = q - (QKH - 1) + i;
            if (qr >= 0) {
                unsigned lds_off = (unsigned)(size_t)(const void*)&rows[i][8];
                tdm_load_row(lds_off, sc + (size_t)qr * SEQ);
            }
        }
        __builtin_amdgcn_s_wait_tensorcnt(0);
    }
    __syncthreads();

    for (int k = tid; k < SEQ; k += 256) {
        float a = 0.f;
#pragma unroll
        for (int i = 0; i < QKH; i++)
#pragma unroll
            for (int j = 0; j < KKW; j++)
                a += kk[i * KKW + j] * rows[i][k + 3 + j];   // col (k-5+j)+8
        conv[k] = a;
    }
    __syncthreads();
    float m = -INFINITY;
    for (int k = tid; k <= q; k += 256) m = fmaxf(m, conv[k]);
    red[tid] = m; __syncthreads();
    for (int s = 128; s > 0; s >>= 1) { if (tid < s) red[tid] = fmaxf(red[tid], red[tid + s]); __syncthreads(); }
    m = red[0]; __syncthreads();
    float sum = 0.f;
    for (int k = tid; k <= q; k += 256) { float e = __expf(conv[k] - m); conv[k] = e; sum += e; }
    red[tid] = sum; __syncthreads();
    for (int s = 128; s > 0; s >>= 1) { if (tid < s) red[tid] += red[tid + s]; __syncthreads(); }
    float inv = 1.f / red[0];
    for (int k = tid; k < SEQ; k += 256)
        probs[(size_t)q * SEQ + k] = tobf((k <= q) ? conv[k] * inv : 0.f);
}

// ---------------- RMS sub-norm + sigmoid gate, one wave per (s,h) ----------------
__global__ __launch_bounds__(256) void norm_gate_kernel(
    const float* __restrict__ attn, const float* __restrict__ slw,
    const float* __restrict__ gw, const float* __restrict__ gb,
    __bf16* __restrict__ attnb)
{
    int wid  = (int)((blockIdx.x * blockDim.x + threadIdx.x) >> 5);
    int lane = threadIdx.x & 31;
    int s = wid >> 4, h = wid & 15;
    if (s >= SEQ) return;
    const float* row = attn + (size_t)s * DIMN + h * HD;
    float x0 = row[lane], x1 = row[lane + 32];
    float ss = x0 * x0 + x1 * x1;
    for (int o = 16; o > 0; o >>= 1) ss += __shfl_xor(ss, o, 32);
    float r = rsqrtf(ss * (1.f / HD) + EPSV);
    float y0 = x0 * r * slw[lane], y1 = x1 * r * slw[lane + 32];
    float g = y0 * gw[lane] + y1 * gw[lane + 32];
    for (int o = 16; o > 0; o >>= 1) g += __shfl_xor(g, o, 32);
    g += gb[0];
    g = 1.f / (1.f + __expf(-g));
    size_t d = (size_t)s * DIMN + h * HD;
    attnb[d + lane]      = tobf(y0 * g);
    attnb[d + lane + 32] = tobf(y1 * g);
}

// ---------------- host orchestration ----------------
extern "C" void kernel_launch(void* const* d_in, const int* in_sizes, int n_in,
                              void* d_out, int out_size, void* d_ws, size_t ws_size,
                              hipStream_t stream)
{
    const float* x    = (const float*)d_in[0];
    const float* freq = (const float*)d_in[1];
    const float* wq   = (const float*)d_in[2];
    const float* wk   = (const float*)d_in[3];
    const float* wv   = (const float*)d_in[4];
    const float* wo   = (const float*)d_in[5];
    const float* mta  = (const float*)d_in[6];
    const float* slw  = (const float*)d_in[7];
    const float* gw   = (const float*)d_in[8];
    const float* gb   = (const float*)d_in[9];
    float* out = (float*)d_out;

    char* base = (char*)d_ws; size_t off = 0;
    auto alloc = [&](size_t bytes) -> void* {
        void* p = base + off; off += (bytes + 255) & ~(size_t)255; return p;
    };
    __bf16* xb   = (__bf16*)alloc((size_t)SEQ * DIMN * 2);
    __bf16* wqt  = (__bf16*)alloc((size_t)DIMN * DIMN * 2);
    __bf16* wkt  = (__bf16*)alloc((size_t)DIMN * DIMN * 2);
    __bf16* wvt  = (__bf16*)alloc((size_t)DIMN * DIMN * 2);
    __bf16* wot  = (__bf16*)alloc((size_t)DIMN * DIMN * 2);
    float*  qf   = (float*) alloc((size_t)SEQ * DIMN * 4);
    float*  kf   = (float*) alloc((size_t)SEQ * DIMN * 4);
    float*  vf   = (float*) alloc((size_t)SEQ * DIMN * 4);
    __bf16* qb   = (__bf16*)alloc((size_t)HEADS * SEQ * HD * 2);
    __bf16* kb   = (__bf16*)alloc((size_t)HEADS * SEQ * HD * 2);
    __bf16* vt   = (__bf16*)alloc((size_t)HEADS * HD * SEQ * 2);
    float*  sc   = (float*) alloc((size_t)SEQ * SEQ * 4);
    __bf16* pb   = (__bf16*)alloc((size_t)SEQ * SEQ * 2);
    float*  attn = (float*) alloc((size_t)SEQ * DIMN * 4);
    __bf16* attb = (__bf16*)alloc((size_t)SEQ * DIMN * 2);

    cvt_bf16_kernel<<<(SEQ * DIMN) / 256, 256, 0, stream>>>(x, xb, SEQ * DIMN);
    transpose_bf16_kernel<<<(DIMN * DIMN) / 256, 256, 0, stream>>>(wq, wqt, DIMN, DIMN);
    transpose_bf16_kernel<<<(DIMN * DIMN) / 256, 256, 0, stream>>>(wk, wkt, DIMN, DIMN);
    transpose_bf16_kernel<<<(DIMN * DIMN) / 256, 256, 0, stream>>>(wv, wvt, DIMN, DIMN);
    transpose_bf16_kernel<<<(DIMN * DIMN) / 256, 256, 0, stream>>>(wo, wot, DIMN, DIMN);

    // 32x64 tile per wave -> (M/32)*(N/64) waves, 8 waves per block
    int gemm_blocks = (SEQ / 32) * (DIMN / 64) / 8;          // 128
    gemm_bf16_nt<<<gemm_blocks, 256, 0, stream>>>(xb, DIMN, wqt, DIMN, qf, DIMN, SEQ, DIMN, DIMN);
    gemm_bf16_nt<<<gemm_blocks, 256, 0, stream>>>(xb, DIMN, wkt, DIMN, kf, DIMN, SEQ, DIMN, DIMN);
    gemm_bf16_nt<<<gemm_blocks, 256, 0, stream>>>(xb, DIMN, wvt, DIMN, vf, DIMN, SEQ, DIMN, DIMN);

    rope_pack_kernel<<<(SEQ * HEADS * 32) / 256, 256, 0, stream>>>(qf, kf, vf, freq, qb, kb, vt);

    float scale = 0.125f;  // rsqrt(64)
    for (int h = 0; h < HEADS; ++h) {
        const __bf16* qh = qb + (size_t)h * SEQ * HD;
        const __bf16* kh = kb + (size_t)h * SEQ * HD;
        const __bf16* vh = vt + (size_t)h * HD * SEQ;
        scores_kernel<<<(SEQ / 16) * (SEQ / 64) / 8, 256, 0, stream>>>(qh, kh, sc, scale);
        conv_softmax_kernel<<<SEQ, 256, 0, stream>>>(sc, mta + (size_t)h * QKH * KKW, pb);
        gemm_bf16_nt<<<(SEQ / 32) * (HD / 64) / 8, 256, 0, stream>>>(
            pb, SEQ, vh, SEQ, attn + (size_t)h * HD, DIMN, SEQ, HD, SEQ);
    }

    norm_gate_kernel<<<(SEQ * HEADS * 32) / 256, 256, 0, stream>>>(attn, slw, gw, gb, attb);
    gemm_bf16_nt<<<gemm_blocks, 256, 0, stream>>>(attb, DIMN, wot, DIMN, out, DIMN, SEQ, DIMN, DIMN);
}
